// StochasticLSTMCell_16776142258523
// MI455X (gfx1250) — matrix operand order repaired and checked by reading, not compiled, not run on hardware
//
#include <hip/hip_runtime.h>
#include <hip/hip_bf16.h>

#define T_STEPS 2048
#define BATCH   32
#define IDIM    256
#define HDIM    256
#define NWG     16
#define HC      16                 // hidden units per workgroup
#define NCOLS   64                 // 4 gates * HC
#define NT      256                // threads per block (8 wave32)
#define TS_PER_BLK 16              // timesteps per precompute block

#define AS1 __attribute__((address_space(1)))
#define AS3 __attribute__((address_space(3)))

#if __has_builtin(__builtin_amdgcn_global_load_async_to_lds_b128) && \
    __has_builtin(__builtin_amdgcn_s_wait_asynccnt)
#define HAVE_ASYNC_LDS 1
#else
#define HAVE_ASYNC_LDS 0
#endif

typedef __attribute__((ext_vector_type(16))) __bf16 bf16x16;
typedef __attribute__((ext_vector_type(8)))  float  f32x8;
typedef __attribute__((ext_vector_type(4)))  float  f32x4;
typedef __attribute__((ext_vector_type(4)))  int    i32x4;

union FragAB { bf16x16 v; f32x4 q[2]; };

struct SM {
  __bf16 WT[IDIM * NCOLS];      // 32 KB : W slice, transposed  WT[k][n]
  __bf16 UT[HDIM * NCOLS];      // 32 KB : U slice, transposed  UT[k][n]
  __bf16 xdrop[BATCH * IDIM];   // 16 KB : x_t * zx  (bf16)   [fallback path]
  __bf16 hdrop[BATCH * HDIM];   // 16 KB : h_{t-1} * zh (bf16)
  float  gates[BATCH * NCOLS];  //  8 KB
  float  c[BATCH * HC];         //  2 KB
  float  bias[NCOLS];
  float  zx[IDIM];
  float  zh[HC];
};

__device__ __forceinline__ void grid_barrier(int* arrive, int* gen) {
  __syncthreads();
  if (threadIdx.x == 0) {
    __threadfence();
    int g = __hip_atomic_load(gen, __ATOMIC_ACQUIRE, __HIP_MEMORY_SCOPE_AGENT);
    int a = __hip_atomic_fetch_add(arrive, 1, __ATOMIC_ACQ_REL, __HIP_MEMORY_SCOPE_AGENT);
    if (a == NWG - 1) {
      __hip_atomic_store(arrive, 0, __ATOMIC_RELAXED, __HIP_MEMORY_SCOPE_AGENT);
      __hip_atomic_fetch_add(gen, 1, __ATOMIC_ACQ_REL, __HIP_MEMORY_SCOPE_AGENT);
    } else {
      while (__hip_atomic_load(gen, __ATOMIC_ACQUIRE, __HIP_MEMORY_SCOPE_AGENT) == g) {
        __builtin_amdgcn_s_sleep(1);
      }
    }
    __threadfence();
  }
  __syncthreads();
}

__device__ __forceinline__ float sigmoidf_fast(float v) {
  return 1.0f / (1.0f + __expf(-v));
}

// ---------------------------------------------------------------------------
// Parallel precompute:  P[t, b, wg, n] = (x_t * zx) @ W_slice^T   (raw, no bias)
// Grid: (T/TS_PER_BLK) * NWG blocks.  Saturates the chip; removes the input
// projection (8 WMMAs + 32KB x-stage) from the sequential critical path.
// ---------------------------------------------------------------------------
__global__ void __launch_bounds__(NT, 1)
xw_precompute(const float* __restrict__ x,    // [T,B,I]
              const float* __restrict__ zx,   // [I]
              const float* __restrict__ W,    // [4H,I]
              float* __restrict__ P)          // [T,B,NWG,NCOLS]
{
  __shared__ struct {
    __bf16 WT[IDIM * NCOLS];
    __bf16 xd[BATCH * IDIM];
    float  zx[IDIM];
  } sp;
  const int tid = threadIdx.x;
  const int wg  = blockIdx.x & (NWG - 1);
  const int t0  = (blockIdx.x >> 4) * TS_PER_BLK;
  const int hstart = wg * HC;

  for (int idx = tid; idx < IDIM * NCOLS; idx += NT) {
    int n = idx & (NCOLS - 1);
    int k = idx >> 6;
    int r = (n >> 4) * HDIM + hstart + (n & (HC - 1));
    sp.WT[idx] = (__bf16)W[(size_t)r * IDIM + k];
  }
  if (tid < IDIM) sp.zx[tid] = zx[tid];
  __syncthreads();

  const int lane  = tid & 31;
  const int wave  = tid >> 5;
  const int mi    = wave >> 2;
  const int ni    = wave & 3;
  const int half  = lane >> 4;
  const int koff8 = half * 8;
  const int mrow  = mi * 16 + (lane & 15);
  const int nbase = ni * 16 + (lane & 15);
  const int mbase = mi * 16 + 8 * half;

  for (int t = t0; t < t0 + TS_PER_BLK; ++t) {
    const float* xg = x + (size_t)t * (BATCH * IDIM);
    for (int idx = tid; idx < BATCH * IDIM; idx += NT) {
      int k = idx & (IDIM - 1);
      sp.xd[idx] = (__bf16)(xg[idx] * sp.zx[k]);
    }
    __syncthreads();

    f32x8 acc = {};
#pragma unroll
    for (int kb = 0; kb < IDIM / 32; ++kb) {
      FragAB a, b;
      const __bf16* ap = &sp.xd[mrow * IDIM + kb * 32];
      a.q[0] = *(const f32x4*)(ap + koff8);
      a.q[1] = *(const f32x4*)(ap + 16 + koff8);
      const __bf16* bp = &sp.WT[(kb * 32) * NCOLS + nbase];
      for (int kk = 0; kk < 16; ++kk) b.v[kk] = bp[((kk & 8) * 2 + koff8 + (kk & 7)) * NCOLS];
      acc = __builtin_amdgcn_wmma_f32_16x16x32_bf16(false, a.v, false, b.v,
                                                    (short)0, acc, false, false);
    }
#pragma unroll
    for (int r = 0; r < 8; ++r) {
      P[(((size_t)t * BATCH + (mbase + r)) * NWG + wg) * NCOLS + nbase] = acc[r];
    }
    __syncthreads();
  }
}

// ---------------------------------------------------------------------------
// Persistent recurrent kernel: one WG per 16 hidden units, grid barrier per
// step, bf16 WMMA for hdrop @ U^T accumulating on precomputed x-projection.
// ---------------------------------------------------------------------------
__global__ void __launch_bounds__(NT, 1)
lstm_persistent(const float* __restrict__ x,    // [T,B,I]
                const float* __restrict__ zx,   // [I]
                const float* __restrict__ zh,   // [H]
                const float* __restrict__ W,    // [4H,I]
                const float* __restrict__ bW,   // [4H]
                const float* __restrict__ U,    // [4H,H]
                const float* __restrict__ bU,   // [4H]
                float* __restrict__ out,        // hn[T,B,H] ++ hT[B,H] ++ cT[B,H]
                int* __restrict__ bar,          // [arrive, gen]
                __bf16* __restrict__ hbuf,      // 2 * [B*H] bf16 double buffer
                const float* __restrict__ Pg)   // [T,B,NWG,NCOLS] or nullptr
{
  __shared__ SM s;
  const int tid    = threadIdx.x;
  const int wg     = blockIdx.x;
  const int hstart = wg * HC;
  const size_t TBH = (size_t)T_STEPS * BATCH * HDIM;

  // ---- one-time init ----
  for (int idx = tid; idx < IDIM * NCOLS; idx += NT) {
    int n = idx & (NCOLS - 1);
    int k = idx >> 6;
    int r = (n >> 4) * HDIM + hstart + (n & (HC - 1));
    if (!Pg) s.WT[idx] = (__bf16)W[(size_t)r * IDIM + k];
    s.UT[idx] = (__bf16)U[(size_t)r * HDIM + k];
  }
  if (tid < NCOLS) {
    int r = (tid >> 4) * HDIM + hstart + (tid & (HC - 1));
    s.bias[tid] = bW[r] + bU[r];
  }
  if (tid < IDIM) s.zx[tid] = zx[tid];
  if (tid < HC)   s.zh[tid] = zh[hstart + tid];
  for (int e = tid; e < BATCH * HC; e += NT) s.c[e] = 0.0f;

  __bf16* hb1 = hbuf + BATCH * HDIM;            // buffer read at t=0
  for (int e = tid; e < BATCH * HC; e += NT) {
    int b = e >> 4, j = e & (HC - 1);
    hb1[b * HDIM + hstart + j] = (__bf16)0.0f;
  }
  grid_barrier(bar, bar + 1);

  const int lane  = tid & 31;
  const int wave  = tid >> 5;
  const int mi    = wave >> 2;
  const int ni    = wave & 3;
  const int half  = lane >> 4;
  const int koff8 = half * 8;
  const int mrow  = mi * 16 + (lane & 15);
  const int nbase = ni * 16 + (lane & 15);
  const int mbase = mi * 16 + 8 * half;

  for (int t = 0; t < T_STEPS; ++t) {
    // ---- stage hdrop_{t-1} into LDS (async DMA path on gfx1250) ----
    const __bf16* hr = hbuf + ((t + 1) & 1) * (BATCH * HDIM);
#if HAVE_ASYNC_LDS
    {
      AS3 char* lbase = (AS3 char*)&s.hdrop[0];
      const AS1 char* gbase = (const AS1 char*)hr;
#pragma unroll
      for (int i = 0; i < (BATCH * HDIM * 2) / (16 * NT); ++i) {   // 4 x b128
        int off = (tid + i * NT) * 16;
        __builtin_amdgcn_global_load_async_to_lds_b128(
            (AS1 i32x4*)(gbase + off), (AS3 i32x4*)(lbase + off), 0, 0);
      }
    }
#else
    for (int idx = tid; idx < (BATCH * HDIM) / 2; idx += NT) {
      ((unsigned int*)s.hdrop)[idx] = ((const unsigned int*)hr)[idx];
    }
#endif
    if (!Pg) {
      const float* xg = x + (size_t)t * (BATCH * IDIM);
      for (int idx = tid; idx < BATCH * IDIM; idx += NT) {
        int k = idx & (IDIM - 1);
        s.xdrop[idx] = (__bf16)(xg[idx] * s.zx[k]);
      }
      if (t + 1 < T_STEPS) __builtin_prefetch(xg + BATCH * IDIM + tid * 32, 0, 1);
    } else if (t + 1 < T_STEPS) {
      int row = tid & 31, coloff = (tid >> 5) * 8;
      __builtin_prefetch(
          &Pg[(((size_t)(t + 1) * BATCH + row) * NWG + wg) * NCOLS + coloff], 0, 1);
    }
#if HAVE_ASYNC_LDS
    __builtin_amdgcn_s_wait_asynccnt(0);
#endif
    __syncthreads();

    // ---- gates tile [32 x 64]: init from precomputed x@W^T, add h@U^T ----
    f32x8 acc = {};
    if (Pg) {
#pragma unroll
      for (int r = 0; r < 8; ++r) {
        acc[r] = Pg[(((size_t)t * BATCH + (mbase + r)) * NWG + wg) * NCOLS + nbase];
      }
    } else {
#pragma unroll
      for (int kb = 0; kb < IDIM / 32; ++kb) {
        FragAB a, b;
        const __bf16* ap = &s.xdrop[mrow * IDIM + kb * 32];
        a.q[0] = *(const f32x4*)(ap + koff8);
        a.q[1] = *(const f32x4*)(ap + 16 + koff8);
        const __bf16* bp = &s.WT[(kb * 32) * NCOLS + nbase];
        for (int kk = 0; kk < 16; ++kk) b.v[kk] = bp[((kk & 8) * 2 + koff8 + (kk & 7)) * NCOLS];
        acc = __builtin_amdgcn_wmma_f32_16x16x32_bf16(false, a.v, false, b.v,
                                                      (short)0, acc, false, false);
      }
    }
#pragma unroll
    for (int kb = 0; kb < HDIM / 32; ++kb) {
      FragAB a, b;
      const __bf16* ap = &s.hdrop[mrow * HDIM + kb * 32];
      a.q[0] = *(const f32x4*)(ap + koff8);
      a.q[1] = *(const f32x4*)(ap + 16 + koff8);
      const __bf16* bp = &s.UT[(kb * 32) * NCOLS + nbase];
      for (int kk = 0; kk < 16; ++kk) b.v[kk] = bp[((kk & 8) * 2 + koff8 + (kk & 7)) * NCOLS];
      acc = __builtin_amdgcn_wmma_f32_16x16x32_bf16(false, a.v, false, b.v,
                                                    (short)0, acc, false, false);
    }
#pragma unroll
    for (int r = 0; r < 8; ++r) {
      s.gates[(mbase + r) * NCOLS + nbase] = acc[r];
    }
    __syncthreads();

    // ---- elementwise LSTM cell update for our 16 hidden units ----
    __bf16* hw = hbuf + (t & 1) * (BATCH * HDIM);
    for (int e = tid; e < BATCH * HC; e += NT) {
      int b = e >> 4, j = e & (HC - 1);
      const float* gr = &s.gates[b * NCOLS];
      float iv = sigmoidf_fast(gr[j]      + s.bias[j]);
      float fv = sigmoidf_fast(gr[16 + j] + s.bias[16 + j]);
      float ov = sigmoidf_fast(gr[32 + j] + s.bias[32 + j]);
      float gv = tanhf(gr[48 + j] + s.bias[48 + j]);
      float cn = fv * s.c[e] + iv * gv;
      s.c[e] = cn;
      float hn = ov * tanhf(cn);
      out[((size_t)t * BATCH + b) * HDIM + hstart + j] = hn;
      hw[b * HDIM + hstart + j] = (__bf16)(hn * s.zh[j]);
      if (t == T_STEPS - 1) {
        out[TBH + (size_t)b * HDIM + hstart + j] = hn;                        // h_T
        out[TBH + (size_t)BATCH * HDIM + (size_t)b * HDIM + hstart + j] = cn; // c_T
      }
    }
    grid_barrier(bar, bar + 1);
  }
}

__global__ void init_barrier(int* p) {
  if (threadIdx.x < 2) p[threadIdx.x] = 0;
}

extern "C" void kernel_launch(void* const* d_in, const int* in_sizes, int n_in,
                              void* d_out, int out_size, void* d_ws, size_t ws_size,
                              hipStream_t stream) {
  (void)in_sizes; (void)n_in; (void)out_size;
  const float* x  = (const float*)d_in[0];
  const float* zx = (const float*)d_in[1];
  const float* zh = (const float*)d_in[2];
  const float* W  = (const float*)d_in[3];
  const float* bW = (const float*)d_in[4];
  const float* U  = (const float*)d_in[5];
  const float* bU = (const float*)d_in[6];
  float* out = (float*)d_out;

  int*    bar  = (int*)d_ws;
  __bf16* hbuf = (__bf16*)((char*)d_ws + 128);     // 2 * 16 KB double buffer

  // Precomputed input projections P[T,B,NWG,NCOLS] (f32) if workspace allows.
  const size_t P_OFF   = 65536;
  const size_t P_BYTES = (size_t)T_STEPS * BATCH * NWG * NCOLS * sizeof(float);
  float* P = (ws_size >= P_OFF + P_BYTES) ? (float*)((char*)d_ws + P_OFF) : nullptr;

  init_barrier<<<1, 64, 0, stream>>>(bar);
  if (P) {
    xw_precompute<<<(T_STEPS / TS_PER_BLK) * NWG, NT, 0, stream>>>(x, zx, W, P);
  }
  lstm_persistent<<<NWG, NT, 0, stream>>>(x, zx, zh, W, bW, U, bU, out, bar, hbuf, P);
}
